// ScaledDotProductAttention_42125039239585
// MI455X (gfx1250) — compile-verified, matrix operand order checked by reading, not run on hardware
//
#include <hip/hip_runtime.h>

typedef _Float16 v16h  __attribute__((ext_vector_type(16)));
typedef _Float16 h4    __attribute__((ext_vector_type(4)));
typedef __fp16   f16x2 __attribute__((ext_vector_type(2)));   // cvt_pkrtz return type
typedef float    v8f   __attribute__((ext_vector_type(8)));

#define S_DIM 2048
#define D_DIM 64
#define H_DIM 8
#define KV_CHUNK 64
#define QTILE 64        // Q rows per block (4 waves x 16 rows)
#define NTHREADS 128

union H4u  { h4   v; f16x2 h[2]; };
union V16u { v16h v; f16x2 h[8]; };

__global__ __launch_bounds__(NTHREADS)
void fa_wmma_kernel(const float* __restrict__ Q,
                    const float* __restrict__ K,
                    const float* __restrict__ V,
                    const int*   __restrict__ Mask,
                    float* __restrict__ Out)
{
    // LDS: K chunk (row major, f16), V chunk (transposed, f16), P scratch per wave
    __shared__ __align__(128) _Float16 Kt[KV_CHUNK][D_DIM];      // [kv][d]   8 KB
    __shared__ __align__(128) _Float16 VtT[D_DIM][KV_CHUNK];     // [d][kv]   8 KB
    __shared__ __align__(128) _Float16 Pbuf[4][16][KV_CHUNK];    // per-wave  8 KB

    const int tid  = threadIdx.x;
    const int wave = tid >> 5;
    const int lane = tid & 31;
    const int half = lane >> 4;     // 0 or 1 (lane groups per WMMA layout)
    const int lm   = lane & 15;     // lane % 16

    const int bh = blockIdx.y;              // b*H + h
    const int b  = bh / H_DIM;
    const size_t base = (size_t)bh * S_DIM * D_DIM;

    const float* Qb = Q + base;
    const float* Kb = K + base;
    const float* Vb = V + base;
    const int*   Mb = Mask + (size_t)b * S_DIM * S_DIM;  // head-broadcast mask

    const int q0   = blockIdx.x * QTILE + wave * 16;  // first Q row of this wave
    const int qrow = q0 + lm;                          // A-fragment row for this lane

    // ---- Load Q A-fragments (16x32 f16 each, D split into two K=32 chunks) ----
    // A layout: lane L holds row M=L%16, k = (L/16)*16 + e, e=0..15 (contiguous in d)
    // Fold (1/sqrt(64)) * log2(e) into Q: softmax runs in the log2 domain.
    v16h aq[2];
    {
        const float scale = 0.18033688011112042f;  // 0.125 * log2(e)
        const float* qp = Qb + (size_t)qrow * D_DIM + half * 16;
        #pragma unroll
        for (int c = 0; c < 2; ++c) {
            const float* p = qp + c * 32;
            V16u a;
            #pragma unroll
            for (int i = 0; i < 8; ++i)
                a.h[i] = __builtin_amdgcn_cvt_pkrtz(p[2*i] * scale, p[2*i+1] * scale);
            aq[c] = a.v;
        }
    }

    // All-ones B fragment: P x ones = per-row sums, replicated across lanes (C layout)
    v16h ones;
    #pragma unroll
    for (int e = 0; e < 16; ++e) ones[e] = (_Float16)1.0f;

    // Online-softmax state (log2 domain): rows m = 8*half + v, replicated per half
    float m_i[8];
    v8f   lacc;     // running denominator, accumulated through WMMA C operand
    v8f   o[4];
    #pragma unroll
    for (int v = 0; v < 8; ++v) m_i[v] = -3.0e38f;
    lacc = (v8f){0,0,0,0,0,0,0,0};
    #pragma unroll
    for (int d = 0; d < 4; ++d) o[d] = (v8f){0,0,0,0,0,0,0,0};

    for (int kv0 = 0; kv0 < S_DIM; kv0 += KV_CHUNK) {
        __syncthreads();  // previous chunk fully consumed

        // ---- Cooperative stage: 4x4 register-transpose blocks, packed cvt ----
        #pragma unroll
        for (int i = 0; i < 2; ++i) {
            int t2 = tid + i * NTHREADS;      // 4x4 tile id, 0..255
            int tr = t2 >> 4;                 // kv block (0..15)
            int tc = t2 & 15;                 // d  block (0..15)
            const float* kp = Kb + (size_t)(kv0 + tr*4) * D_DIM + tc*4;
            const float* vp = Vb + (size_t)(kv0 + tr*4) * D_DIM + tc*4;
            float4 kf[4], vf[4];
            #pragma unroll
            for (int r = 0; r < 4; ++r) {
                kf[r] = *(const float4*)(kp + r * D_DIM);
                vf[r] = *(const float4*)(vp + r * D_DIM);
            }
            #pragma unroll
            for (int r = 0; r < 4; ++r) {
                H4u kk;
                kk.h[0] = __builtin_amdgcn_cvt_pkrtz(kf[r].x, kf[r].y);
                kk.h[1] = __builtin_amdgcn_cvt_pkrtz(kf[r].z, kf[r].w);
                *(h4*)&Kt[tr*4 + r][tc*4] = kk.v;   // ds_store_b64
            }
            #pragma unroll
            for (int c = 0; c < 4; ++c) {
                H4u vv;
                vv.h[0] = __builtin_amdgcn_cvt_pkrtz(((const float*)&vf[0])[c],
                                                     ((const float*)&vf[1])[c]);
                vv.h[1] = __builtin_amdgcn_cvt_pkrtz(((const float*)&vf[2])[c],
                                                     ((const float*)&vf[3])[c]);
                *(h4*)&VtT[tc*4 + c][tr*4] = vv.v;  // ds_store_b64
            }
        }
        if (kv0 + KV_CHUNK < S_DIM) {
            __builtin_prefetch(Kb + (size_t)(kv0 + KV_CHUNK) * D_DIM, 0, 1);
            __builtin_prefetch(Vb + (size_t)(kv0 + KV_CHUNK) * D_DIM, 0, 1);
        }

        // ---- Mask bias tiles (issued before the barrier; independent of LDS) ----
        // C layout: lane holds column n = lm, rows m = 8*half + v.
        // cb[j][v] = mask ? -1e9 : 0, fed as the WMMA C operand (no post-WMMA select).
        const int* mp = Mb + (size_t)(q0 + 8*half) * S_DIM + kv0 + lm;
        int mki[8][4];
        #pragma unroll
        for (int v = 0; v < 8; ++v)
            #pragma unroll
            for (int j = 0; j < 4; ++j)
                mki[v][j] = mp[(size_t)v * S_DIM + j * 16];
        v8f cb[4];
        #pragma unroll
        for (int j = 0; j < 4; ++j)
            #pragma unroll
            for (int v = 0; v < 8; ++v)
                cb[j][v] = (float)mki[v][j] * -1.0e9f;

        __syncthreads();

        // ---- Score tiles: S = Q K^T + mask_bias (log2 domain) ----
        v8f sc[4];
        #pragma unroll
        for (int j = 0; j < 4; ++j) {
            v16h bk0 = *(const v16h*)&Kt[j*16 + lm][0  + half*16];
            v16h bk1 = *(const v16h*)&Kt[j*16 + lm][32 + half*16];
            v8f s = __builtin_amdgcn_wmma_f32_16x16x32_f16(false, aq[0], false, bk0,
                                                           (short)0, cb[j], false, false);
            s = __builtin_amdgcn_wmma_f32_16x16x32_f16(false, aq[1], false, bk1,
                                                       (short)0, s, false, false);
            sc[j] = s;
        }

        // ---- One combined online-softmax update per 64-wide chunk ----
        #pragma unroll
        for (int v = 0; v < 8; ++v) {
            float s0 = sc[0][v], s1 = sc[1][v], s2 = sc[2][v], s3 = sc[3][v];
            float rm = fmaxf(fmaxf(s0, s1), fmaxf(s2, s3));
            rm = fmaxf(rm, __shfl_xor(rm, 1, 32));
            rm = fmaxf(rm, __shfl_xor(rm, 2, 32));
            rm = fmaxf(rm, __shfl_xor(rm, 4, 32));
            rm = fmaxf(rm, __shfl_xor(rm, 8, 32));

            float mnew  = fmaxf(m_i[v], rm);
            float alpha = __builtin_exp2f(m_i[v] - mnew);
            m_i[v] = mnew;
            lacc[v] *= alpha;
            #pragma unroll
            for (int d = 0; d < 4; ++d) o[d][v] *= alpha;

            // exp2(-1e9-ish - mnew) underflows to exactly 0 => masked P is 0
            Pbuf[wave][8*half + v][ 0 + lm] = (_Float16)__builtin_exp2f(s0 - mnew);
            Pbuf[wave][8*half + v][16 + lm] = (_Float16)__builtin_exp2f(s1 - mnew);
            Pbuf[wave][8*half + v][32 + lm] = (_Float16)__builtin_exp2f(s2 - mnew);
            Pbuf[wave][8*half + v][48 + lm] = (_Float16)__builtin_exp2f(s3 - mnew);
        }

        // Wave-private LDS store->load of P: enforce DS ordering explicitly
        asm volatile("s_wait_dscnt 0" ::: "memory");

        v16h ap0 = *(const v16h*)&Pbuf[wave][lm][0  + half*16];  // kv 0..31
        v16h ap1 = *(const v16h*)&Pbuf[wave][lm][32 + half*16];  // kv 32..63

        // l += rowsum(P), accumulated entirely inside the matrix pipe
        lacc = __builtin_amdgcn_wmma_f32_16x16x32_f16(false, ap0, false, ones,
                                                      (short)0, lacc, false, false);
        lacc = __builtin_amdgcn_wmma_f32_16x16x32_f16(false, ap1, false, ones,
                                                      (short)0, lacc, false, false);

        // ---- O += P V : B[k][n] = V[kv0+k][d*16+n] from transposed V tile ----
        #pragma unroll
        for (int d = 0; d < 4; ++d) {
            v16h bv0 = *(const v16h*)&VtT[d*16 + lm][0  + half*16];
            v16h bv1 = *(const v16h*)&VtT[d*16 + lm][32 + half*16];
            o[d] = __builtin_amdgcn_wmma_f32_16x16x32_f16(false, ap0, false, bv0,
                                                          (short)0, o[d], false, false);
            o[d] = __builtin_amdgcn_wmma_f32_16x16x32_f16(false, ap1, false, bv1,
                                                          (short)0, o[d], false, false);
        }
    }

    // ---- Epilogue: O / l, store fp32 [B,H,S,D] ----
    float* op = Out + base + (size_t)(q0 + 8*half) * D_DIM;
    #pragma unroll
    for (int v = 0; v < 8; ++v) {
        float inv = (lacc[v] > 0.0f) ? (1.0f / lacc[v]) : 0.0f;
        #pragma unroll
        for (int d = 0; d < 4; ++d)
            op[(size_t)v * D_DIM + d*16 + lm] = o[d][v] * inv;
    }
}

extern "C" void kernel_launch(void* const* d_in, const int* in_sizes, int n_in,
                              void* d_out, int out_size, void* d_ws, size_t ws_size,
                              hipStream_t stream) {
    (void)in_sizes; (void)n_in; (void)out_size; (void)d_ws; (void)ws_size;
    const float* Q    = (const float*)d_in[0];
    const float* K    = (const float*)d_in[1];
    const float* V    = (const float*)d_in[2];
    const int*   Mask = (const int*)d_in[3];
    float* Out = (float*)d_out;

    const int B = 4, H = H_DIM;
    dim3 grid(S_DIM / QTILE, B * H);   // (32, 32) blocks
    dim3 block(NTHREADS);
    fa_wmma_kernel<<<grid, block, 0, stream>>>(Q, K, V, Mask, Out);
}
